// Attention_65094524338392
// MI455X (gfx1250) — compile-verified, matrix operand order checked by reading, not run on hardware
//
#include <hip/hip_runtime.h>
#include <hip/hip_bf16.h>

// ---------------- problem constants ----------------
#define S_LEN 2048
#define DIMK  2048
#define NH    32
#define NKV   8
#define NREP  4
#define HD    64
#define SCALE_F 0.125f      // 1/sqrt(64)
#define NEGINF (-1.0e30f)

typedef __bf16 bf16_t;
typedef __attribute__((ext_vector_type(16))) __bf16 v16bf;
typedef __attribute__((ext_vector_type(8)))  __bf16 v8bf;
typedef __attribute__((ext_vector_type(8)))  float  v8f;

__device__ __forceinline__ bf16_t f2bf(float f) {
  unsigned u = __builtin_bit_cast(unsigned, f);
  unsigned r = (u + 0x7fffu + ((u >> 16) & 1u)) >> 16;   // round-to-nearest-even
  return __builtin_bit_cast(bf16_t, (unsigned short)r);
}

// Load one 16x32 bf16 A/B fragment for v_wmma_f32_16x16x32_bf16 from a
// row-major matrix (rows over M or N, columns over K).
// ISA layout: lanes 0-15 own rows 0-15, K halves split between lane halves:
//   elements 0..7  -> K = k0 + khalf + j
//   elements 8..15 -> K = k0 + 16 + khalf + j   (khalf = 8 for lanes 16-31)
// => two aligned 16-byte global_load_b128 per lane.
__device__ __forceinline__ v16bf load_frag_g(const bf16_t* __restrict__ base,
                                             int row0, int stride, int k0) {
  int lane = threadIdx.x & 31;
  const bf16_t* p = base + (size_t)(row0 + (lane & 15)) * stride + k0 + ((lane >> 4) << 3);
  v8bf lo = *(const v8bf*)(p);
  v8bf hi = *(const v8bf*)(p + 16);
  v16bf r;
#pragma unroll
  for (int j = 0; j < 8; ++j) { r[j] = lo[j]; r[j + 8] = hi[j]; }
  return r;
}

#define WMMA_BF16(A, B, C) \
  __builtin_amdgcn_wmma_f32_16x16x32_bf16(false, (A), false, (B), (short)0, (C), false, false)

// ---------------- fp32 -> bf16 conversion ----------------
__global__ void f32_to_bf16_kernel(const float* __restrict__ src,
                                   bf16_t* __restrict__ dst, long n) {
  long i = (long)blockIdx.x * blockDim.x + threadIdx.x;
  if (i < n) dst[i] = f2bf(src[i]);
}

// ---------------- generic bf16 WMMA GEMM: C[M,N] = A[M,K] * W[N,K]^T + bias ----------------
// one wave per 16x128 output strip: 8 independent WMMAs per A fragment
__global__ __launch_bounds__(32) void gemm_bf16_wmma(
    const bf16_t* __restrict__ A, const bf16_t* __restrict__ W,
    const float* __restrict__ bias, float* __restrict__ C,
    int M, int N, int K) {
  (void)M;
  const int tiles_n = N >> 7;
  const int m0 = (blockIdx.x / tiles_n) << 4;
  const int n0 = (blockIdx.x % tiles_n) << 7;
  const int lane = threadIdx.x;

  v8f acc[8];
#pragma unroll
  for (int t = 0; t < 8; ++t) acc[t] = v8f{};

  for (int k0 = 0; k0 < K; k0 += 32) {
    // prefetch next K-chunk of A into cache (global_prefetch_b8)
    if (k0 + 32 < K)
      __builtin_prefetch(A + (size_t)(m0 + (lane & 15)) * K + k0 + 32, 0, 1);
    v16bf a = load_frag_g(A, m0, K, k0);
#pragma unroll
    for (int t = 0; t < 8; ++t) {
      v16bf b = load_frag_g(W, n0 + t * 16, K, k0);   // col n of W^T == row n of W
      acc[t] = WMMA_BF16(a, b, acc[t]);
    }
  }
  // D layout: VGPR r -> (M = r + 8*(lane>=16), N = lane&15)
  const int mo = m0 + ((lane >> 4) << 3);
  const int nc = lane & 15;
#pragma unroll
  for (int t = 0; t < 8; ++t)
#pragma unroll
    for (int r = 0; r < 8; ++r) {
      int row = mo + r, col = n0 + t * 16 + nc;
      C[(size_t)row * N + col] = acc[t][r] + bias[col];
    }
}

// ---------------- RoPE + pack to head-major bf16 (and V transposed) ----------------
__global__ void rope_pack_kernel(const float* __restrict__ qf, const float* __restrict__ kf,
                                 const float* __restrict__ vf, const float* __restrict__ rope,
                                 bf16_t* __restrict__ qb, bf16_t* __restrict__ kb,
                                 bf16_t* __restrict__ vT) {
  long idx = (long)blockIdx.x * blockDim.x + threadIdx.x;  // S * NH * (HD/2)
  if (idx >= (long)S_LEN * NH * (HD / 2)) return;
  int d = (int)(idx & 31);
  int h = (int)((idx >> 5) & (NH - 1));
  int s = (int)(idx >> 10);
  float c  = rope[(size_t)s * HD + d];
  float sn = rope[(size_t)s * HD + 32 + d];

  float x1 = qf[(size_t)s * (NH * HD) + h * HD + d];
  float x2 = qf[(size_t)s * (NH * HD) + h * HD + d + 32];
  qb[((size_t)h * S_LEN + s) * HD + d]      = f2bf(x1 * c - x2 * sn);
  qb[((size_t)h * S_LEN + s) * HD + d + 32] = f2bf(x1 * sn + x2 * c);

  if (h < NKV) {
    float k1 = kf[(size_t)s * (NKV * HD) + h * HD + d];
    float k2 = kf[(size_t)s * (NKV * HD) + h * HD + d + 32];
    kb[((size_t)h * S_LEN + s) * HD + d]      = f2bf(k1 * c - k2 * sn);
    kb[((size_t)h * S_LEN + s) * HD + d + 32] = f2bf(k1 * sn + k2 * c);
    float v1 = vf[(size_t)s * (NKV * HD) + h * HD + d];
    float v2 = vf[(size_t)s * (NKV * HD) + h * HD + d + 32];
    vT[((size_t)h * HD + d) * S_LEN + s]      = f2bf(v1);
    vT[((size_t)h * HD + d + 32) * S_LEN + s] = f2bf(v2);
  }
}

// ---------------- flash attention: one wave per (head, 16-row q strip) ----------------
// consecutive blocks sweep adjacent q strips of the SAME head -> that head's
// K/V (256 KB bf16) stays hot in L2 across the whole causal sweep.
__global__ __launch_bounds__(32) void attn_kernel(
    const bf16_t* __restrict__ qb, const bf16_t* __restrict__ kb,
    const bf16_t* __restrict__ vT, const float* __restrict__ sinks,
    bf16_t* __restrict__ attn_out) {
  __shared__ bf16_t sP[16 * 32];   // P re-layout tile (D -> A fragment)

  const int qtiles = S_LEN / 16;
  const int h  = blockIdx.x / qtiles;
  const int q0 = (blockIdx.x % qtiles) << 4;
  const int hkv = h / NREP;
  const int lane = threadIdx.x;
  const int hi8 = (lane >> 4) << 3;
  const int nc  = lane & 15;

  const bf16_t* Q  = qb + (size_t)h   * S_LEN * HD;
  const bf16_t* Km = kb + (size_t)hkv * S_LEN * HD;
  const bf16_t* Vt = vT + (size_t)hkv * HD * S_LEN;

  // Q A-fragments for the two K-chunks of HD=64
  v16bf aq0 = load_frag_g(Q, q0, HD, 0);
  v16bf aq1 = load_frag_g(Q, q0, HD, 32);

  float m_r[8], l_r[8];
#pragma unroll
  for (int r = 0; r < 8; ++r) { m_r[r] = -3.0e30f; l_r[r] = 0.0f; }
  v8f acc[4] = {v8f{}, v8f{}, v8f{}, v8f{}};

  const int kv_end = q0 + 16;                 // causal: last needed col = q0+15
  for (int kv0 = 0; kv0 < kv_end; kv0 += 32) {
    // prefetch next KV tile's K rows (one row per lane)
    if (kv0 + 32 < kv_end)
      __builtin_prefetch(Km + (size_t)(kv0 + 32 + lane) * HD, 0, 1);
    // ---- scores: 16q x 32kv (two 16x16 tiles, K = HD = 64) ----
    v8f s01[2];
#pragma unroll
    for (int t = 0; t < 2; ++t) {
      v16bf bk0 = load_frag_g(Km, kv0 + t * 16, HD, 0);
      v16bf bk1 = load_frag_g(Km, kv0 + t * 16, HD, 32);
      v8f sc{};
      sc = WMMA_BF16(aq0, bk0, sc);
      sc = WMMA_BF16(aq1, bk1, sc);
      s01[t] = sc;
    }
    // ---- online softmax (row = q0 + r + hi8, reductions within 16-lane halves) ----
#pragma unroll
    for (int r = 0; r < 8; ++r) {
      const int row = q0 + r + hi8;
      const int c0  = kv0 + nc;
      float s0 = s01[0][r] * SCALE_F; if (c0 > row)      s0 = NEGINF;
      float s1 = s01[1][r] * SCALE_F; if (c0 + 16 > row) s1 = NEGINF;
      float mx = fmaxf(s0, s1);
      mx = fmaxf(mx, __shfl_xor(mx, 1));
      mx = fmaxf(mx, __shfl_xor(mx, 2));
      mx = fmaxf(mx, __shfl_xor(mx, 4));
      mx = fmaxf(mx, __shfl_xor(mx, 8));
      const float mnew = fmaxf(m_r[r], mx);
      const float p0 = __expf(s0 - mnew);
      const float p1 = __expf(s1 - mnew);
      sP[(r + hi8) * 32 + nc]      = f2bf(p0);
      sP[(r + hi8) * 32 + 16 + nc] = f2bf(p1);
      float ps = p0 + p1;
      ps += __shfl_xor(ps, 1);
      ps += __shfl_xor(ps, 2);
      ps += __shfl_xor(ps, 4);
      ps += __shfl_xor(ps, 8);
      const float ef = __expf(m_r[r] - mnew);
      l_r[r] = l_r[r] * ef + ps;
      m_r[r] = mnew;
#pragma unroll
      for (int t = 0; t < 4; ++t) acc[t][r] *= ef;
    }
    __syncthreads();
    // ---- reload P as A-fragment from LDS (two ds_load_b128 per lane) ----
    v16bf ap;
    {
      const bf16_t* p = &sP[(lane & 15) * 32 + ((lane >> 4) << 3)];
      v8bf lo = *(const v8bf*)(p);
      v8bf hi = *(const v8bf*)(p + 16);
#pragma unroll
      for (int j = 0; j < 8; ++j) { ap[j] = lo[j]; ap[j + 8] = hi[j]; }
    }
    __syncthreads();
    // ---- O += P(16x32) * V(32x64): B-fragments contiguous from V^T ----
#pragma unroll
    for (int t = 0; t < 4; ++t) {
      v16bf bv = load_frag_g(Vt, t * 16, S_LEN, kv0);
      acc[t] = WMMA_BF16(ap, bv, acc[t]);
    }
  }

  // ---- epilogue: lse, sigmoid sink scale, normalize, store bf16 ----
  const float sink = sinks[h];
#pragma unroll
  for (int r = 0; r < 8; ++r) {
    const float lse = m_r[r] + __logf(l_r[r]);
    const float ss  = 1.0f / (1.0f + __expf(sink - lse));   // sigmoid(lse - sink)
    const float inv = ss / l_r[r];
    const int row = q0 + r + hi8;
#pragma unroll
    for (int t = 0; t < 4; ++t)
      attn_out[(size_t)row * (NH * HD) + h * HD + t * 16 + nc] = f2bf(acc[t][r] * inv);
  }
}

// ---------------- host launcher ----------------
extern "C" void kernel_launch(void* const* d_in, const int* in_sizes, int n_in,
                              void* d_out, int out_size, void* d_ws, size_t ws_size,
                              hipStream_t stream) {
  (void)in_sizes; (void)n_in; (void)out_size; (void)ws_size;
  const float* x     = (const float*)d_in[0];
  const float* rope  = (const float*)d_in[1];
  const float* wq_w  = (const float*)d_in[2];
  const float* wq_b  = (const float*)d_in[3];
  const float* wk_w  = (const float*)d_in[4];
  const float* wk_b  = (const float*)d_in[5];
  const float* wv_w  = (const float*)d_in[6];
  const float* wv_b  = (const float*)d_in[7];
  const float* wo_w  = (const float*)d_in[8];
  const float* wo_b  = (const float*)d_in[9];
  const float* sinks = (const float*)d_in[10];
  float* out = (float*)d_out;

  char* ws = (char*)d_ws;
  size_t off = 0;
  auto alloc = [&](size_t bytes) -> char* {
    char* p = ws + off;
    off += (bytes + 255) & ~(size_t)255;
    return p;
  };
  bf16_t* xb   = (bf16_t*)alloc((size_t)S_LEN * DIMK * 2);
  bf16_t* wqb  = (bf16_t*)alloc((size_t)NH  * HD * DIMK * 2);
  bf16_t* wkb  = (bf16_t*)alloc((size_t)NKV * HD * DIMK * 2);
  bf16_t* wvb  = (bf16_t*)alloc((size_t)NKV * HD * DIMK * 2);
  bf16_t* wob  = (bf16_t*)alloc((size_t)DIMK * NH * HD * 2);
  float*  qf   = (float*) alloc((size_t)S_LEN * NH  * HD * 4);
  float*  kf   = (float*) alloc((size_t)S_LEN * NKV * HD * 4);
  float*  vf   = (float*) alloc((size_t)S_LEN * NKV * HD * 4);
  bf16_t* qb   = (bf16_t*)alloc((size_t)NH  * S_LEN * HD * 2);
  bf16_t* kb   = (bf16_t*)alloc((size_t)NKV * S_LEN * HD * 2);
  bf16_t* vTb  = (bf16_t*)alloc((size_t)NKV * HD * S_LEN * 2);
  bf16_t* attb = (bf16_t*)alloc((size_t)S_LEN * NH * HD * 2);

  // 1) convert activations + weights to bf16
  {
    long n;
    n = (long)S_LEN * DIMK;
    f32_to_bf16_kernel<<<(unsigned)((n + 255) / 256), 256, 0, stream>>>(x, xb, n);
    n = (long)NH * HD * DIMK;
    f32_to_bf16_kernel<<<(unsigned)((n + 255) / 256), 256, 0, stream>>>(wq_w, wqb, n);
    n = (long)NKV * HD * DIMK;
    f32_to_bf16_kernel<<<(unsigned)((n + 255) / 256), 256, 0, stream>>>(wk_w, wkb, n);
    f32_to_bf16_kernel<<<(unsigned)((n + 255) / 256), 256, 0, stream>>>(wv_w, wvb, n);
    n = (long)DIMK * NH * HD;
    f32_to_bf16_kernel<<<(unsigned)((n + 255) / 256), 256, 0, stream>>>(wo_w, wob, n);
  }

  // 2) QKV projections (WMMA GEMMs, fp32 out)
  gemm_bf16_wmma<<<(S_LEN / 16) * ((NH * HD) / 128), 32, 0, stream>>>(
      xb, wqb, wq_b, qf, S_LEN, NH * HD, DIMK);
  gemm_bf16_wmma<<<(S_LEN / 16) * ((NKV * HD) / 128), 32, 0, stream>>>(
      xb, wkb, wk_b, kf, S_LEN, NKV * HD, DIMK);
  gemm_bf16_wmma<<<(S_LEN / 16) * ((NKV * HD) / 128), 32, 0, stream>>>(
      xb, wvb, wv_b, vf, S_LEN, NKV * HD, DIMK);

  // 3) RoPE + pack to head-major bf16 (V transposed)
  {
    long n = (long)S_LEN * NH * (HD / 2);
    rope_pack_kernel<<<(unsigned)((n + 255) / 256), 256, 0, stream>>>(
        qf, kf, vf, rope, qb, kb, vTb);
  }

  // 4) flash attention with sinks
  attn_kernel<<<NH * (S_LEN / 16), 32, 0, stream>>>(qb, kb, vTb, sinks, attb);

  // 5) output projection -> fp32 result
  gemm_bf16_wmma<<<(S_LEN / 16) * (DIMK / 128), 32, 0, stream>>>(
      attb, wob, wo_b, out, S_LEN, DIMK, NH * HD);
}
